// RNNSearchDecoder_50414326120714
// MI455X (gfx1250) — matrix operand order repaired and checked by reading, not compiled, not run on hardware
//
#include <hip/hip_runtime.h>
#include <hip/hip_bf16.h>
#include <stdint.h>

typedef __attribute__((ext_vector_type(16))) __bf16 v16bf;
typedef __attribute__((ext_vector_type(8)))  float  v8f;
typedef unsigned short u16;

#define TT 64
#define BB 128
#define SS 64
#define HH 1024
#define EE 512
#define H2 2048
#define KCAT 3584   /* E + H + 2H */
#define NGATE 2048  /* r rows (H) + candidate rows (H); unused update gate dropped */

__device__ __forceinline__ u16 f32_to_bf16_rne(float f) {
  unsigned u = __float_as_uint(f);
  unsigned r = u + 0x7FFFu + ((u >> 16) & 1u);
  return (u16)(r >> 16);
}

union Frag { uint4 q[2]; v16bf v; };

// ---------------------------------------------------------------------------
// bf16 WMMA GEMM: out[M,N] = A[M,K] * W[N,K]^T + bias[N]
// A row-major with leading dim lda (elements); W row-major [N,K] (weight layout
// as given by the reference). One wave = one 16x16 tile, 4 waves/block along N.
// Fragment element->K mapping follows CDNA5 ISA 7.12.2 (16-bit A 16x32, B 32x16).
// ---------------------------------------------------------------------------
__global__ void gemm_bf16(const u16* __restrict__ A, int lda,
                          const u16* __restrict__ W, int K,
                          const float* __restrict__ bias,
                          float* __restrict__ out, int ldo) {
  const int lane = threadIdx.x & 31;
  const int wave = threadIdx.x >> 5;
  const int half = lane >> 4;
  const int l16  = lane & 15;
  const int mt = blockIdx.y;
  const int nt = blockIdx.x * 4 + wave;
  const u16* arow = A + (size_t)(mt * 16 + l16) * lda;
  const u16* wrow = W + (size_t)(nt * 16 + l16) * K;
  v8f acc = {};
  for (int k0 = 0; k0 < K; k0 += 32) {
    Frag a, b;
    // A: elems 0..7 <- K = k0+8*half .. +7 ; elems 8..15 <- K = k0+16+8*half ..
    a.q[0] = *(const uint4*)(arow + k0 + half * 8);
    a.q[1] = *(const uint4*)(arow + k0 + 16 + half * 8);
    // B: column n = l16 (= row of W), elems 0..7 <- K = k0+8*half .. +7 ; elems 8..15 <- K = k0+16+8*half ..
    b.q[0] = *(const uint4*)(wrow + k0 + half * 8);
    b.q[1] = *(const uint4*)(wrow + k0 + 16 + half * 8);
    acc = __builtin_amdgcn_wmma_f32_16x16x32_bf16(false, a.v, false, b.v,
                                                  (short)0, acc, false, false);
  }
  const int n = nt * 16 + l16;
  const float bv = bias[n];
#pragma unroll
  for (int r = 0; r < 8; ++r) {
    int m = mt * 16 + half * 8 + r;   // C layout: VGPR r -> M = r + 8*half
    out[(size_t)m * ldo + n] = acc[r] + bv;
  }
}

// ----------------------------- prep kernels --------------------------------
__global__ void k_cvt_bf16(const float* __restrict__ s, u16* __restrict__ d, int n) {
  int i = blockIdx.x * blockDim.x + threadIdx.x;
  if (i < n) d[i] = f32_to_bf16_rne(s[i]);
}

__global__ void k_gather_embed(const int* __restrict__ tgt, const float* __restrict__ embed,
                               u16* __restrict__ embT) {
  int i = blockIdx.x * blockDim.x + threadIdx.x;   // T*B*E
  if (i >= TT * BB * EE) return;
  int e = i & (EE - 1);
  int tb = i >> 9;
  int tok = tgt[tb];
  embT[i] = f32_to_bf16_rne(embed[(size_t)tok * EE + e]);
}

// concat [Wig | Whg | Wcg] (r rows only) on top of [Wii | Whi | Wci] -> [2048, 3584]
__global__ void k_build_gate_w(const float* __restrict__ Wig, const float* __restrict__ Whg,
                               const float* __restrict__ Wcg, const float* __restrict__ Wii,
                               const float* __restrict__ Whi, const float* __restrict__ Wci,
                               u16* __restrict__ Wg) {
  int i = blockIdx.x * blockDim.x + threadIdx.x;
  if (i >= NGATE * KCAT) return;
  int k = i % KCAT, o = i / KCAT;
  float val;
  if (o < HH) {
    val = (k < EE)      ? Wig[(size_t)o * EE + k]
        : (k < EE + HH) ? Whg[(size_t)o * HH + (k - EE)]
                        : Wcg[(size_t)o * H2 + (k - EE - HH)];
  } else {
    int oo = o - HH;
    val = (k < EE)      ? Wii[(size_t)oo * EE + k]
        : (k < EE + HH) ? Whi[(size_t)oo * HH + (k - EE)]
                        : Wci[(size_t)oo * H2 + (k - EE - HH)];
  }
  Wg[i] = f32_to_bf16_rne(val);
}

// concat [My | Mh | Mc] -> [2048, 3584] matching x = [y, h_new, ctx]
__global__ void k_build_max_w(const float* __restrict__ My, const float* __restrict__ Mh,
                              const float* __restrict__ Mc, u16* __restrict__ Wm) {
  int i = blockIdx.x * blockDim.x + threadIdx.x;
  if (i >= H2 * KCAT) return;
  int k = i % KCAT, o = i / KCAT;
  float val = (k < EE)      ? My[(size_t)o * EE + k]
            : (k < EE + HH) ? Mh[(size_t)o * HH + (k - EE)]
                            : Mc[(size_t)o * H2 + (k - EE - HH)];
  Wm[i] = f32_to_bf16_rne(val);
}

__global__ void k_build_biases(const float* big, const float* bhg, const float* bcg,
                               const float* bii, const float* bhi, const float* bci,
                               const float* bmy, const float* bmh, const float* bmc,
                               float* __restrict__ bgate, float* __restrict__ bmax) {
  int i = blockIdx.x * blockDim.x + threadIdx.x;
  if (i >= H2) return;
  bmax[i] = bmy[i] + bmh[i] + bmc[i];
  bgate[i] = (i < HH) ? (big[i] + bhg[i] + bcg[i])
                      : (bii[i - HH] + bhi[i - HH] + bci[i - HH]);
}

__global__ void k_init_h(const float* __restrict__ hidden0, float* __restrict__ h,
                         u16* __restrict__ xcat) {
  int i = blockIdx.x * blockDim.x + threadIdx.x;
  if (i >= BB * HH) return;
  int b = i >> 10, o = i & (HH - 1);
  float v = hidden0[i];
  h[i] = v;
  xcat[b * KCAT + EE + o] = f32_to_bf16_rne(v);
}

// ----------------------------- per-step kernels ----------------------------
__global__ void k_set_y(const u16* __restrict__ embT_t, u16* __restrict__ xcat) {
  int i = blockIdx.x * blockDim.x + threadIdx.x;   // B*E
  if (i >= BB * EE) return;
  int b = i >> 9, e = i & (EE - 1);
  xcat[b * KCAT + e] = embT_t[i];
}

// e[s,b] = bv + v . tanh(enc_lin[s,b,:] + q[b,:]) ; one wave per (s,b)
__global__ void k_attn_score(const float* __restrict__ enc_lin, const float* __restrict__ q,
                             const float* __restrict__ v, const float* __restrict__ bv,
                             float* __restrict__ e) {
  int row  = blockIdx.x * 4 + (threadIdx.x >> 5);  // row = s*B + b
  int lane = threadIdx.x & 31;
  int b = row & (BB - 1);
  const float* er = enc_lin + (size_t)row * HH;
  const float* qr = q + (size_t)b * HH;
  float acc = 0.f;
  for (int hI = lane; hI < HH; hI += 32)
    acc += v[hI] * tanhf(er[hI] + qr[hI]);
  for (int off = 16; off > 0; off >>= 1)
    acc += __shfl_down(acc, off, 32);
  if (lane == 0) e[row] = acc + bv[0];
}

// softmax over S per batch column; also writes attns[t,b,s]
__global__ void k_softmax(const float* __restrict__ e, float* __restrict__ alpha,
                          float* __restrict__ attn_t) {
  __shared__ float red[SS];
  int b = blockIdx.x;
  int s = threadIdx.x;                 // 64 threads
  float val = e[s * BB + b];
  red[s] = val; __syncthreads();
  for (int off = SS / 2; off > 0; off >>= 1) {
    if (s < off) red[s] = fmaxf(red[s], red[s + off]);
    __syncthreads();
  }
  float mx = red[0]; __syncthreads();
  float ex = __expf(val - mx);
  red[s] = ex; __syncthreads();
  for (int off = SS / 2; off > 0; off >>= 1) {
    if (s < off) red[s] += red[s + off];
    __syncthreads();
  }
  float a = ex / red[0];
  alpha[s * BB + b] = a;
  attn_t[b * SS + s] = a;
}

// ctx[b,:] = sum_s alpha[s,b] * membank[s,b,:]; writes bf16 straight into xcat
__global__ void k_ctx(const float* __restrict__ alpha, const float* __restrict__ membank,
                      u16* __restrict__ xcat) {
  __shared__ float al[SS];
  int b = blockIdx.x;
  if (threadIdx.x < SS) al[threadIdx.x] = alpha[threadIdx.x * BB + b];
  __syncthreads();
  for (int d = threadIdx.x; d < H2; d += blockDim.x) {
    float acc = 0.f;
    for (int s = 0; s < SS; ++s)
      acc += al[s] * membank[((size_t)s * BB + b) * H2 + d];
    xcat[b * KCAT + EE + HH + d] = f32_to_bf16_rne(acc);
  }
}

// h_new = (1-r)*h + r*tanh(cand); update fp32 carry and bf16 slot in xcat
__global__ void k_gru(const float* __restrict__ gpre, float* __restrict__ h,
                      u16* __restrict__ xcat) {
  int i = blockIdx.x * blockDim.x + threadIdx.x;
  if (i >= BB * HH) return;
  int b = i >> 10, o = i & (HH - 1);
  float r    = 1.f / (1.f + __expf(-gpre[(size_t)b * NGATE + o]));
  float cand = tanhf(gpre[(size_t)b * NGATE + HH + o]);
  float hn = (1.f - r) * h[i] + r * cand;
  h[i] = hn;
  xcat[b * KCAT + EE + o] = f32_to_bf16_rne(hn);
}

__global__ void k_maxout(const float* __restrict__ m, float* __restrict__ dec_t) {
  int i = blockIdx.x * blockDim.x + threadIdx.x;
  if (i >= BB * HH) return;
  int b = i >> 10, o = i & (HH - 1);
  dec_t[i] = fmaxf(m[(size_t)b * H2 + 2 * o], m[(size_t)b * H2 + 2 * o + 1]);
}

__global__ void k_copy(const float* __restrict__ s, float* __restrict__ d, int n) {
  int i = blockIdx.x * blockDim.x + threadIdx.x;
  if (i < n) d[i] = s[i];
}

// ---------------------------------------------------------------------------
extern "C" void kernel_launch(void* const* d_in, const int* in_sizes, int n_in,
                              void* d_out, int out_size, void* d_ws, size_t ws_size,
                              hipStream_t stream) {
  (void)in_sizes; (void)n_in; (void)out_size; (void)ws_size;
  const int*   tgt     = (const int*)  d_in[0];
  const float* membank = (const float*)d_in[1];
  const float* hidden0 = (const float*)d_in[2];
  const float* embed   = (const float*)d_in[3];
  const float* We = (const float*)d_in[4],  *be  = (const float*)d_in[5];
  const float* Ws = (const float*)d_in[6],  *bs  = (const float*)d_in[7];
  const float* v  = (const float*)d_in[8],  *bv  = (const float*)d_in[9];
  const float* Wig= (const float*)d_in[10], *big = (const float*)d_in[11];
  const float* Whg= (const float*)d_in[12], *bhg = (const float*)d_in[13];
  const float* Wcg= (const float*)d_in[14], *bcg = (const float*)d_in[15];
  const float* Wii= (const float*)d_in[16], *bii = (const float*)d_in[17];
  const float* Whi= (const float*)d_in[18], *bhi = (const float*)d_in[19];
  const float* Wci= (const float*)d_in[20], *bci = (const float*)d_in[21];
  const float* Mh = (const float*)d_in[22], *bmh = (const float*)d_in[23];
  const float* My = (const float*)d_in[24], *bmy = (const float*)d_in[25];
  const float* Mc = (const float*)d_in[26], *bmc = (const float*)d_in[27];
  (void)bmh; (void)bmy; (void)bmc; // consumed via k_build_biases below

  char* wsp = (char*)d_ws;
  auto alloc = [&](size_t bytes) {
    char* p = wsp; wsp += (bytes + 255) & ~(size_t)255; return p;
  };
  u16*   mb_bf = (u16*)  alloc((size_t)SS * BB * H2 * 2);
  u16*   We_bf = (u16*)  alloc((size_t)HH * H2 * 2);
  u16*   Ws_bf = (u16*)  alloc((size_t)HH * HH * 2);
  u16*   Wg_bf = (u16*)  alloc((size_t)NGATE * KCAT * 2);
  u16*   Wm_bf = (u16*)  alloc((size_t)H2 * KCAT * 2);
  u16*   embT  = (u16*)  alloc((size_t)TT * BB * EE * 2);
  float* enc   = (float*)alloc((size_t)SS * BB * HH * 4);
  u16*   xcat  = (u16*)  alloc((size_t)BB * KCAT * 2);
  float* hbuf  = (float*)alloc((size_t)BB * HH * 4);
  float* qbuf  = (float*)alloc((size_t)BB * HH * 4);
  float* ebuf  = (float*)alloc((size_t)SS * BB * 4);
  float* abuf  = (float*)alloc((size_t)SS * BB * 4);
  float* gpre  = (float*)alloc((size_t)BB * NGATE * 4);
  float* mbuf  = (float*)alloc((size_t)BB * H2 * 4);
  float* bgate = (float*)alloc((size_t)NGATE * 4);
  float* bmax  = (float*)alloc((size_t)H2 * 4);

  float* dec_h = (float*)d_out;                       // [T,B,H]
  float* h_fin = dec_h + (size_t)TT * BB * HH;        // [B,H]
  float* attns = h_fin + (size_t)BB * HH;             // [T,B,S]

  const int TPB = 256;
  auto nb = [](long n, int b) { return (unsigned)((n + b - 1) / b); };

  // ---- prep (once per call) ----
  k_cvt_bf16<<<nb((long)SS * BB * H2, TPB), TPB, 0, stream>>>(membank, mb_bf, SS * BB * H2);
  k_cvt_bf16<<<nb((long)HH * H2, TPB), TPB, 0, stream>>>(We, We_bf, HH * H2);
  k_cvt_bf16<<<nb((long)HH * HH, TPB), TPB, 0, stream>>>(Ws, Ws_bf, HH * HH);
  k_build_gate_w<<<nb((long)NGATE * KCAT, TPB), TPB, 0, stream>>>(Wig, Whg, Wcg, Wii, Whi, Wci, Wg_bf);
  k_build_max_w<<<nb((long)H2 * KCAT, TPB), TPB, 0, stream>>>(My, Mh, Mc, Wm_bf);
  k_build_biases<<<nb(H2, TPB), TPB, 0, stream>>>(big, bhg, bcg, bii, bhi, bci,
                                                  (const float*)d_in[25], (const float*)d_in[23],
                                                  (const float*)d_in[27], bgate, bmax);
  k_gather_embed<<<nb((long)TT * BB * EE, TPB), TPB, 0, stream>>>(tgt, embed, embT);
  k_init_h<<<nb((long)BB * HH, TPB), TPB, 0, stream>>>(hidden0, hbuf, xcat);

  // enc_lin = membank @ We^T + be  : M=S*B=8192, K=2H, N=H
  {
    dim3 g(HH / 64, (SS * BB) / 16);
    gemm_bf16<<<g, 128, 0, stream>>>(mb_bf, H2, We_bf, H2, be, enc, HH);
  }

  // ---- recurrent loop ----
  for (int t = 0; t < TT; ++t) {
    k_set_y<<<nb((long)BB * EE, TPB), TPB, 0, stream>>>(embT + (size_t)t * BB * EE, xcat);
    { // q = h @ Ws^T + bs (A = h slot of xcat, lda = KCAT)
      dim3 g(HH / 64, BB / 16);
      gemm_bf16<<<g, 128, 0, stream>>>(xcat + EE, KCAT, Ws_bf, HH, bs, qbuf, HH);
    }
    k_attn_score<<<(SS * BB) / 4, 128, 0, stream>>>(enc, qbuf, v, bv, ebuf);
    k_softmax<<<BB, SS, 0, stream>>>(ebuf, abuf, attns + (size_t)t * BB * SS);
    k_ctx<<<BB, 256, 0, stream>>>(abuf, membank, xcat);
    { // fused gate+candidate preactivations: [B, 2048] over K=3584
      dim3 g(NGATE / 64, BB / 16);
      gemm_bf16<<<g, 128, 0, stream>>>(xcat, KCAT, Wg_bf, KCAT, bgate, gpre, NGATE);
    }
    k_gru<<<nb((long)BB * HH, TPB), TPB, 0, stream>>>(gpre, hbuf, xcat);
    { // maxout preactivations with x = [y, h_new, ctx]
      dim3 g(H2 / 64, BB / 16);
      gemm_bf16<<<g, 128, 0, stream>>>(xcat, KCAT, Wm_bf, KCAT, bmax, mbuf, H2);
    }
    k_maxout<<<nb((long)BB * HH, TPB), TPB, 0, stream>>>(mbuf, dec_h + (size_t)t * BB * HH);
  }
  k_copy<<<nb((long)BB * HH, TPB), TPB, 0, stream>>>(hbuf, h_fin, BB * HH);
}
